// UniTransformer_20976620274298
// MI455X (gfx1250) — matrix operand order repaired and checked by reading, not compiled, not run on hardware
//
#include <hip/hip_runtime.h>
#include <hip/hip_bf16.h>
#include <math.h>

// ---------------------------------------------------------------------------
// CDNA5 (gfx1250) wave32 WMMA implementation of the transformer+LSTM forward.
// Heavy matmuls: v_wmma_f32_16x16x32_f16 (f16 A/B, f32 accumulate).
// Tile staging: GLOBAL_LOAD_ASYNC_TO_LDS (ASYNCcnt) when the toolchain
// declares the builtins, plain LDS stores otherwise.
// ---------------------------------------------------------------------------

typedef __attribute__((ext_vector_type(16))) _Float16 v16h;
typedef __attribute__((ext_vector_type(8)))  float    v8f;
typedef int v4i __attribute__((vector_size(16)));  // matches builtin param type

#define DEVI __device__ __forceinline__

static constexpr int Bsz = 32;       // batch
static constexpr int S   = 512;      // sequence
static constexpr int E   = 256;      // model dim
static constexpr int F   = 128;      // ff dim
static constexpr int HD  = 128;      // head-out hidden
static constexpr int NH  = 8;        // heads
static constexpr int NL  = 6;        // layers
static constexpr int M   = Bsz * S;  // 16384 rows
static constexpr int WIN = 300;
static constexpr int WINP = 320;     // K-padded

// ---- async global->LDS copy (CDNA5), graceful fallback ----
#if __has_builtin(__builtin_amdgcn_global_load_async_to_lds_b128)
#define HAVE_ASYNC_LOAD 1
#endif

#if __has_builtin(__builtin_amdgcn_s_wait_asynccnt)
#define ASYNC_WAIT() __builtin_amdgcn_s_wait_asynccnt(0)
#elif defined(HAVE_ASYNC_LOAD)
#define ASYNC_WAIT() asm volatile("s_wait_asynccnt 0" ::: "memory")
#else
#define ASYNC_WAIT() ((void)0)
#endif

typedef __attribute__((address_space(1))) v4i* gv4i_p;  // global int4*
typedef __attribute__((address_space(3))) v4i* lv4i_p;  // LDS int4*

DEVI void async_copy16(void* lds, const void* g) {
#if defined(HAVE_ASYNC_LOAD)
  // Explicit addrspacecast: flat->global is bit-identity, flat->local keeps
  // the low 32 bits (LDS offset) per the ISA aperture rules.
  __builtin_amdgcn_global_load_async_to_lds_b128((gv4i_p)g, (lv4i_p)lds, 0, 0);
#else
  *(uint4*)lds = *(const uint4*)g;
#endif
}

DEVI v8f wmma16(v16h a, v16h b, v8f c) {
  return __builtin_amdgcn_wmma_f32_16x16x32_f16(
      /*neg_a=*/false, a, /*neg_b=*/false, b,
      /*c_mod=*/(short)0, c, /*reuse_a=*/false, /*reuse_b=*/false);
}

// A fragment: 16x32 f16, ISA 7.12.2 "16-bit A-Matrix 16x32" layout.
DEVI v16h frag_a(const _Float16* __restrict__ p, int stride, int lane) {
  const int m = lane & 15, hi = lane >> 4;
  const _Float16* row = p + m * stride;
  v16h f;
#pragma unroll
  for (int v = 0; v < 8; ++v) {
    const int k = 2 * (v & 3) + ((v >= 4) ? 16 : 0) + (hi ? 8 : 0);
    f[2 * v]     = row[k];
    f[2 * v + 1] = row[k + 1];
  }
  return f;
}

// B fragment: 32x16 f16; p points at Bt[n][k] (n-major), Bt[n][k] = B[k][n].
DEVI v16h frag_b(const _Float16* __restrict__ p, int stride, int lane) {
  const int n = lane & 15, hi = lane >> 4;
  const _Float16* row = p + n * stride;
  v16h f;
#pragma unroll
  for (int v = 0; v < 8; ++v) {
    const int k = 2 * (v & 3) + ((v >= 4) ? 16 : 0) + (hi ? 8 : 0);
    f[2 * v]     = row[k];
    f[2 * v + 1] = row[k + 1];
  }
  return f;
}

DEVI float wred_sum(float v) {
#pragma unroll
  for (int m2 = 16; m2 > 0; m2 >>= 1) v += __shfl_xor(v, m2, 32);
  return v;
}
DEVI float wred_max(float v) {
#pragma unroll
  for (int m2 = 16; m2 > 0; m2 >>= 1) v = fmaxf(v, __shfl_xor(v, m2, 32));
  return v;
}
DEVI float sigmf(float x) { return 1.f / (1.f + expf(-x)); }

// ---------------------------------------------------------------------------
// Prep kernels
// ---------------------------------------------------------------------------
__global__ void k_cvt_h(const float* __restrict__ in, _Float16* __restrict__ out, int n) {
  int i = blockIdx.x * 256 + threadIdx.x;
  if (i < n) out[i] = (_Float16)in[i];
}
// inputs [M][300] -> f16 [M][320] (zero-padded K)
__global__ void k_pad_cols_h(const float* __restrict__ in, _Float16* __restrict__ out,
                             int cin, int cout, int n_out) {
  int i = blockIdx.x * 256 + threadIdx.x;
  if (i < n_out) {
    int r = i / cout, c = i - r * cout;
    out[i] = (c < cin) ? (_Float16)in[r * cin + c] : (_Float16)0.f;
  }
}
// batched transpose+convert: nmat matrices [rows][cols] f32 -> [cols][rows] f16
__global__ void k_transpose_bh(const float* __restrict__ in, _Float16* __restrict__ out,
                               int rows, int cols, int n_total) {
  int i = blockIdx.x * 256 + threadIdx.x;
  if (i < n_total) {
    const int per = rows * cols;
    const int mmat = i / per, rem = i - mmat * per;
    const int r = rem / cols, c = rem - r * cols;
    out[(size_t)mmat * per + c * rows + r] = (_Float16)in[i];
  }
}
// embed_w [300][256] f32 -> Wt f16 [256][320] zero-padded K
__global__ void k_embed_t(const float* __restrict__ in, _Float16* __restrict__ out) {
  int i = blockIdx.x * 256 + threadIdx.x;
  if (i < E * WINP) {
    int n = i / WINP, k = i - n * WINP;
    out[i] = (k < WIN) ? (_Float16)in[k * E + n] : (_Float16)0.f;
  }
}
__global__ void k_addvec(const float* __restrict__ a, const float* __restrict__ b,
                         float* __restrict__ o, int n) {
  int i = blockIdx.x * 256 + threadIdx.x;
  if (i < n) o[i] = a[i] + b[i];
}

// ---------------------------------------------------------------------------
// LayerNorm (E=256): g*(x-mean)/(sqrt(var)+eps)+b, var = sum/(E-1).
// One wave per row, 8 rows per 256-thread block.
// ---------------------------------------------------------------------------
__global__ void k_layernorm256(const float* __restrict__ x, const float* __restrict__ g,
                               const float* __restrict__ bb, _Float16* __restrict__ y) {
  const int wave = threadIdx.x >> 5, lane = threadIdx.x & 31;
  const int row = blockIdx.x * 8 + wave;
  const float* xr = x + (size_t)row * E;
  float v[8], s = 0.f, s2 = 0.f;
#pragma unroll
  for (int i = 0; i < 8; ++i) {
    float t = xr[lane + i * 32];
    v[i] = t; s += t; s2 += t * t;
  }
  s = wred_sum(s); s2 = wred_sum(s2);
  const float mean = s * (1.f / 256.f);
  float var = fmaxf((s2 - 256.f * mean * mean) * (1.f / 255.f), 0.f);
  const float inv = 1.f / (sqrtf(var) + 1e-6f);
  _Float16* yr = y + (size_t)row * E;
#pragma unroll
  for (int i = 0; i < 8; ++i) {
    int c = lane + i * 32;
    yr[c] = (_Float16)(g[c] * (v[i] - mean) * inv + bb[c]);
  }
}

// ---------------------------------------------------------------------------
// WMMA GEMM: D[M x N] = op( A_f16[M x K] @ Wt_f16[N x K]^T + bias )
// Workgroup tile 128x128, 8 waves as 2(M) x 4(N); wave tile 64x32 = 4x2 WMMA
// tiles (8 wmma / K-step). Double-buffered LDS with async global->LDS staging.
// FLAGS: 1=ReLU, 2=residual add from R (f32), 4=store f32, 8=store f16.
// ---------------------------------------------------------------------------
template <int FLAGS>
__global__ void gemm16(const _Float16* __restrict__ A, int lda,
                       const _Float16* __restrict__ Wt,  // [N][K]
                       const float* __restrict__ bias,
                       const float* __restrict__ R,
                       float* __restrict__ Cf, _Float16* __restrict__ Ch,
                       int ldc, int K) {
  __shared__ _Float16 As[2][128 * 40];
  __shared__ _Float16 Bs[2][128 * 40];
  const int tid = threadIdx.x;
  const int wave = tid >> 5, lane = tid & 31;
  const int wm = wave & 1, wn = wave >> 1;   // 2 x 4
  const int m0 = blockIdx.x * 128, n0 = blockIdx.y * 128;
  const int hi = lane >> 4, nl = lane & 15;

  v8f acc[4][2];
#pragma unroll
  for (int mt = 0; mt < 4; ++mt)
#pragma unroll
    for (int nt = 0; nt < 2; ++nt) {
      const float bv = bias[n0 + wn * 32 + nt * 16 + nl];
#pragma unroll
      for (int r = 0; r < 8; ++r) acc[mt][nt][r] = bv;
    }

  // Stage one 128x32 A tile and 128x32 Bt tile (16B per lane per issue).
  auto stage = [&](int buf, int k0) {
#pragma unroll
    for (int i = 0; i < 2; ++i) {
      const int idx = tid + i * 256;
      const int row = idx >> 2, seg = idx & 3;
      async_copy16(&As[buf][row * 40 + seg * 8],
                   &A[(size_t)(m0 + row) * lda + k0 + seg * 8]);
      async_copy16(&Bs[buf][row * 40 + seg * 8],
                   &Wt[(size_t)(n0 + row) * K + k0 + seg * 8]);
    }
  };

  const int nk = K >> 5;
  stage(0, 0);
  for (int kc = 0; kc < nk; ++kc) {
    const int cur = kc & 1;
    ASYNC_WAIT();        // this wave's copies into `cur` are resident
    __syncthreads();     // => everyone's copies into `cur` are resident
    if (kc + 1 < nk) stage(cur ^ 1, (kc + 1) * 32);

    v16h af[4], bf[2];
#pragma unroll
    for (int mt = 0; mt < 4; ++mt)
      af[mt] = frag_a(&As[cur][(wm * 64 + mt * 16) * 40], 40, lane);
#pragma unroll
    for (int nt = 0; nt < 2; ++nt)
      bf[nt] = frag_b(&Bs[cur][(wn * 32 + nt * 16) * 40], 40, lane);
#pragma unroll
    for (int mt = 0; mt < 4; ++mt)
#pragma unroll
      for (int nt = 0; nt < 2; ++nt)
        acc[mt][nt] = wmma16(af[mt], bf[nt], acc[mt][nt]);
    __syncthreads();     // protect `cur` from being restaged while in use
  }

#pragma unroll
  for (int mt = 0; mt < 4; ++mt)
#pragma unroll
    for (int nt = 0; nt < 2; ++nt) {
      const int nc = n0 + wn * 32 + nt * 16 + nl;
#pragma unroll
      for (int r = 0; r < 8; ++r) {
        const int mr = m0 + wm * 64 + mt * 16 + r + 8 * hi;
        float vv = acc[mt][nt][r];
        if (FLAGS & 1) vv = fmaxf(vv, 0.f);
        if (FLAGS & 2) vv += R[(size_t)mr * ldc + nc];
        if (FLAGS & 4) Cf[(size_t)mr * ldc + nc] = vv;
        if (FLAGS & 8) Ch[(size_t)mr * ldc + nc] = (_Float16)vv;
      }
    }
}

// ---------------------------------------------------------------------------
// Attention: one workgroup per (b, h, 64-query tile). dk=32 -> one WMMA per
// score tile; full 64x512 score tile in LDS; f32 softmax; P@V via WMMA.
// ---------------------------------------------------------------------------
static constexpr int SC_B = 64 * 520 * 4;
static constexpr int PS_B = 64 * 520 * 2;
static constexpr int QS_B = 64 * 40 * 2;
static constexpr int KS_B = 512 * 40 * 2;
static constexpr int VT_B = 32 * 520 * 2;
static constexpr int ATT_SMEM = SC_B + PS_B + QS_B + KS_B + VT_B;  // 279040 B

__global__ void k_attention(const _Float16* __restrict__ Q, const _Float16* __restrict__ Kx,
                            const _Float16* __restrict__ V, const float* __restrict__ maskp,
                            _Float16* __restrict__ O) {
  extern __shared__ char sm[];
  float*    Sc = (float*)sm;                                   // [64][520]
  _Float16* Ps = (_Float16*)(sm + SC_B);                       // [64][520]
  _Float16* Qs = (_Float16*)(sm + SC_B + PS_B);                // [64][40]
  _Float16* Ks = (_Float16*)(sm + SC_B + PS_B + QS_B);         // [512][40]
  _Float16* Vt = (_Float16*)(sm + SC_B + PS_B + QS_B + KS_B);  // [32][520]

  const int tid = threadIdx.x;
  const int wave = tid >> 5, lane = tid & 31;
  const int bh = blockIdx.x;
  const int b = bh >> 3, h = bh & 7;
  const int q0 = blockIdx.y * 64;
  const int hi = lane >> 4, nl = lane & 15;

  // Stage Q tile (async).
  {
    const int r = tid >> 2, seg = tid & 3;
    async_copy16(&Qs[r * 40 + seg * 8],
                 &Q[((size_t)(b * S + q0 + r)) * E + h * 32 + seg * 8]);
  }
  // Stage all 512 K rows (Bt layout for scores is K's row-major rows; async).
#pragma unroll
  for (int i = 0; i < 8; ++i) {
    const int idx = tid + i * 256;
    const int r = idx >> 2, seg = idx & 3;
    async_copy16(&Ks[r * 40 + seg * 8],
                 &Kx[((size_t)(b * S + r)) * E + h * 32 + seg * 8]);
  }
  // Stage V transposed (scatter, synchronous): Vt[d][key].
#pragma unroll
  for (int i = 0; i < 2; ++i) {
    const int key = tid + i * 256;
    const _Float16* vr = &V[((size_t)(b * S + key)) * E + h * 32];
#pragma unroll
    for (int d = 0; d < 32; ++d) Vt[d * 520 + key] = vr[d];
  }
  ASYNC_WAIT();
  __syncthreads();

  // Scores: 128 tiles over 8 waves; 1 WMMA each (K = dk = 32). Q fragments
  // are loop-invariant -> hoisted.
  const float scale = 0.17677669529663689f;  // 1/sqrt(32)
  v16h af[4];
#pragma unroll
  for (int mt = 0; mt < 4; ++mt) af[mt] = frag_a(&Qs[(mt * 16) * 40], 40, lane);
  for (int i = 0; i < 16; ++i) {
    const int tile = wave * 16 + i;
    const int nt = tile >> 2;
    v16h bf = frag_b(&Ks[(nt * 16) * 40], 40, lane);
    v8f acc = {};
    acc = wmma16(af[tile & 3], bf, acc);
#pragma unroll
    for (int r = 0; r < 8; ++r)
      Sc[((tile & 3) * 16 + r + 8 * hi) * 520 + nt * 16 + nl] = acc[r] * scale;
  }
  __syncthreads();

  // Softmax per row (8 rows/wave). mask[b,q]==0 -> uniform 1/512 (reference
  // masks whole query rows with -1e9).
  for (int rr = 0; rr < 8; ++rr) {
    const int row = wave * 8 + rr;
    const float mk = maskp[b * S + q0 + row];
    float* srow = &Sc[row * 520];
    _Float16* prow = &Ps[row * 520];
    if (mk == 0.f) {
      const _Float16 u = (_Float16)(1.f / 512.f);
      for (int j = lane; j < 512; j += 32) prow[j] = u;
    } else {
      float vals[16];
      float mx = -3.0e38f;
#pragma unroll
      for (int i = 0; i < 16; ++i) {
        vals[i] = srow[i * 32 + lane];
        mx = fmaxf(mx, vals[i]);
      }
      mx = wred_max(mx);
      float sum = 0.f;
#pragma unroll
      for (int i = 0; i < 16; ++i) {
        const float e = expf(vals[i] - mx);
        vals[i] = e; sum += e;
      }
      sum = wred_sum(sum);
      const float inv = 1.f / sum;
#pragma unroll
      for (int i = 0; i < 16; ++i) prow[i * 32 + lane] = (_Float16)(vals[i] * inv);
    }
  }
  __syncthreads();

  // O = P @ V : 4 M-tiles x 2 N-tiles over 8 waves, K = 512 -> 16 chunks.
  {
    const int mt = wave & 3, nt = wave >> 2;
    v8f acc = {};
    for (int kc = 0; kc < 16; ++kc) {
      v16h a = frag_a(&Ps[(mt * 16) * 520 + kc * 32], 520, lane);
      v16h bf = frag_b(&Vt[(nt * 16) * 520 + kc * 32], 520, lane);
      acc = wmma16(a, bf, acc);
    }
#pragma unroll
    for (int r = 0; r < 8; ++r)
      O[((size_t)(b * S + q0 + mt * 16 + r + 8 * hi)) * E + h * 32 + nt * 16 + nl] =
          (_Float16)acc[r];
  }
}

// ---------------------------------------------------------------------------
// LSTM decoder scan: single 1024-thread (32-wave) workgroup, state in LDS.
// gates[32x1024] = [o_prev|enc_t][32x512] @ Wih^T + h[32x256] @ Whh^T
// (Wih's native [1024x512] layout IS the WMMA Bt layout for i @ Wih.T).
// ---------------------------------------------------------------------------
static constexpr int LG_B = 32 * 1032 * 4;  // gates f32
static constexpr int LC_B = 32 * 256 * 4;   // c f32
static constexpr int LI_B = 32 * 520 * 2;   // [o_prev | enc_t] f16
static constexpr int LH_B = 32 * 264 * 2;   // h f16
static constexpr int LD_B = 32 * 136 * 2;   // hidden f16
static constexpr int LSTM_SMEM = LG_B + LC_B + LI_B + LH_B + LD_B;  // 223744 B

__global__ void k_lstm(const _Float16* __restrict__ enc, const float* __restrict__ maskp,
                       const _Float16* __restrict__ Wih, const _Float16* __restrict__ Whh,
                       const float* __restrict__ gbias,
                       const _Float16* __restrict__ W1T, const float* __restrict__ b1,
                       const float* __restrict__ w2, const float* __restrict__ b2,
                       const float* __restrict__ h0, const float* __restrict__ c0,
                       float* __restrict__ out) {
  extern __shared__ char sm[];
  float*    G  = (float*)sm;                                  // [32][1032]
  float*    cS = (float*)(sm + LG_B);                         // [32][256]
  _Float16* iS = (_Float16*)(sm + LG_B + LC_B);               // [32][520]
  _Float16* hS = (_Float16*)(sm + LG_B + LC_B + LI_B);        // [32][264]
  _Float16* hd = (_Float16*)(sm + LG_B + LC_B + LI_B + LH_B); // [32][136]

  const int tid = threadIdx.x;
  const int wave = tid >> 5, lane = tid & 31;
  const int hi = lane >> 4, nl = lane & 15;

  for (int idx = tid; idx < 32 * 256; idx += 1024) {
    const int m = idx >> 8, e = idx & 255;
    cS[m * 256 + e] = c0[e];
    hS[m * 264 + e] = (_Float16)h0[e];
    iS[m * 520 + e] = (_Float16)0.f;
  }
  __syncthreads();

  for (int t = 0; t < S; ++t) {
    // enc_t -> concat input (async copy, 16B per thread).
    {
      const int m = tid >> 5;
      const int c = (tid & 31) * 8;
      async_copy16(&iS[m * 520 + 256 + c], &enc[((size_t)(m * S + t)) * E + c]);
    }
    ASYNC_WAIT();
    __syncthreads();

    // Gates: 2 M-tiles x 64 N-tiles = 128 tiles, 4 per wave; K = 512 + 256.
#pragma unroll
    for (int j = 0; j < 4; ++j) {
      const int tile = wave * 4 + j;
      const int mt = tile & 1, nt = tile >> 1;
      const int ncol = nt * 16 + nl;
      const float bv = gbias[ncol];
      v8f acc;
#pragma unroll
      for (int r = 0; r < 8; ++r) acc[r] = bv;
      for (int kc = 0; kc < 16; ++kc) {
        v16h a = frag_a(&iS[(mt * 16) * 520 + kc * 32], 520, lane);
        v16h bf = frag_b(&Wih[(size_t)(nt * 16) * 512 + kc * 32], 512, lane);
        acc = wmma16(a, bf, acc);
      }
      for (int kc = 0; kc < 8; ++kc) {
        v16h a = frag_a(&hS[(mt * 16) * 264 + kc * 32], 264, lane);
        v16h bf = frag_b(&Whh[(size_t)(nt * 16) * 256 + kc * 32], 256, lane);
        acc = wmma16(a, bf, acc);
      }
#pragma unroll
      for (int r = 0; r < 8; ++r)
        G[(mt * 16 + r + 8 * hi) * 1032 + ncol] = acc[r];
    }
    __syncthreads();

    // LSTM cell elementwise (8 elements per thread).
    {
      const int m = tid >> 5;
      const int e0 = (tid & 31) * 8;
#pragma unroll
      for (int e = e0; e < e0 + 8; ++e) {
        const float gi = G[m * 1032 + e];
        const float gf = G[m * 1032 + 256 + e];
        const float gg = G[m * 1032 + 512 + e];
        const float go = G[m * 1032 + 768 + e];
        const float cn = sigmf(gf) * cS[m * 256 + e] + sigmf(gi) * tanhf(gg);
        const float hn = sigmf(go) * tanhf(cn);
        cS[m * 256 + e] = cn;
        hS[m * 264 + e] = (_Float16)hn;   // h for next step
        iS[m * 520 + e] = (_Float16)hn;   // o_prev for next step
      }
    }
    __syncthreads();

    // hid = relu(h @ out_w1 + b1): 16 tiles on waves 0..15; K = 256.
    if (wave < 16) {
      const int mt = wave & 1, nt = wave >> 1;
      const int ncol = nt * 16 + nl;
      const float bv = b1[ncol];
      v8f acc;
#pragma unroll
      for (int r = 0; r < 8; ++r) acc[r] = bv;
      for (int kc = 0; kc < 8; ++kc) {
        v16h a = frag_a(&hS[(mt * 16) * 264 + kc * 32], 264, lane);
        v16h bf = frag_b(&W1T[(size_t)(nt * 16) * 256 + kc * 32], 256, lane);
        acc = wmma16(a, bf, acc);
      }
#pragma unroll
      for (int r = 0; r < 8; ++r)
        hd[(mt * 16 + r + 8 * hi) * 136 + ncol] = (_Float16)fmaxf(acc[r], 0.f);
    }
    __syncthreads();

    // p = hid @ out_w2 + b2, masked store.
    if (tid < 32) {
      float s = b2[0];
      for (int jj = 0; jj < 128; ++jj) s += (float)hd[tid * 136 + jj] * w2[jj];
      out[tid * S + t] = s * maskp[tid * S + t];
    }
    __syncthreads();
  }
}

// ---------------------------------------------------------------------------
// Host orchestration
// ---------------------------------------------------------------------------
extern "C" void kernel_launch(void* const* d_in, const int* in_sizes, int n_in,
                              void* d_out, int out_size, void* d_ws, size_t ws_size,
                              hipStream_t stream) {
  (void)in_sizes; (void)n_in; (void)out_size; (void)ws_size;
  const float* inputs   = (const float*)d_in[0];
  const float* mask     = (const float*)d_in[1];
  const float* embed_w  = (const float*)d_in[3];
  const float* embed_b  = (const float*)d_in[4];
  const float* qkvo_w   = (const float*)d_in[5];
  const float* qkvo_b   = (const float*)d_in[6];
  const float* ln_g     = (const float*)d_in[7];
  const float* ln_b     = (const float*)d_in[8];
  const float* ff_w1    = (const float*)d_in[9];
  const float* ff_b1    = (const float*)d_in[10];
  const float* ff_w2    = (const float*)d_in[11];
  const float* ff_b2    = (const float*)d_in[12];
  const float* enc_ln_g = (const float*)d_in[13];
  const float* enc_ln_b = (const float*)d_in[14];
  const float* lstm_wih = (const float*)d_in[15];
  const float* lstm_whh = (const float*)d_in[16];
  const float* lstm_bih = (const float*)d_in[17];
  const float* lstm_bhh = (const float*)d_in[18];
  const float* dec_h0   = (const float*)d_in[19];
  const float* dec_c0   = (const float*)d_in[20];
  const float* out_w1   = (const float*)d_in[21];
  const float* out_b1   = (const float*)d_in[22];
  const float* out_w2   = (const float*)d_in[23];
  const float* out_b2   = (const float*)d_in[24];
  float* out = (float*)d_out;

  char* ws = (char*)d_ws;
  size_t off = 0;
  auto alloc = [&](size_t bytes) -> char* {
    char* p = ws + off;
    off = (off + bytes + 255) & ~(size_t)255;
    return p;
  };

  float*    xF     = (float*)alloc((size_t)M * E * 4);        // residual stream f32
  _Float16* yH     = (_Float16*)alloc((size_t)M * E * 2);     // LN output f16
  _Float16* qH     = (_Float16*)alloc((size_t)M * E * 2);
  _Float16* kH     = (_Float16*)alloc((size_t)M * E * 2);
  _Float16* vH     = (_Float16*)alloc((size_t)M * E * 2);
  _Float16* aH     = (_Float16*)alloc((size_t)M * E * 2);     // attention out
  _Float16* fH     = (_Float16*)alloc((size_t)M * F * 2);     // ff hidden
  _Float16* encH   = (_Float16*)alloc((size_t)M * E * 2);     // final LN out
  _Float16* inPadH = (_Float16*)alloc((size_t)M * WINP * 2);  // padded inputs
  _Float16* embWt  = (_Float16*)alloc((size_t)E * WINP * 2);        // [256][320]
  _Float16* qkvoWt = (_Float16*)alloc((size_t)NL * 4 * E * E * 2);  // [n][k]
  _Float16* ffW1t  = (_Float16*)alloc((size_t)NL * F * E * 2);      // [128][256]
  _Float16* ffW2t  = (_Float16*)alloc((size_t)NL * E * F * 2);      // [256][128]
  _Float16* wihH   = (_Float16*)alloc((size_t)1024 * 512 * 2);
  _Float16* whhH   = (_Float16*)alloc((size_t)1024 * 256 * 2);
  _Float16* w1tH   = (_Float16*)alloc((size_t)HD * E * 2);
  float*    lbias  = (float*)alloc(1024 * 4);

  auto grid1 = [](int n) { return dim3((unsigned)((n + 255) / 256)); };

  // ---- prep: pad/convert/transpose (all GEMM weights go to Bt [N][K]) ----
  k_pad_cols_h<<<grid1(M * WINP), 256, 0, stream>>>(inputs, inPadH, WIN, WINP, M * WINP);
  k_embed_t<<<grid1(E * WINP), 256, 0, stream>>>(embed_w, embWt);
  k_transpose_bh<<<grid1(NL * 4 * E * E), 256, 0, stream>>>(qkvo_w, qkvoWt, E, E, NL * 4 * E * E);
  k_transpose_bh<<<grid1(NL * E * F), 256, 0, stream>>>(ff_w1, ffW1t, E, F, NL * E * F);
  k_transpose_bh<<<grid1(NL * F * E), 256, 0, stream>>>(ff_w2, ffW2t, F, E, NL * F * E);
  k_cvt_h<<<grid1(1024 * 512), 256, 0, stream>>>(lstm_wih, wihH, 1024 * 512);
  k_cvt_h<<<grid1(1024 * 256), 256, 0, stream>>>(lstm_whh, whhH, 1024 * 256);
  k_transpose_bh<<<grid1(E * HD), 256, 0, stream>>>(out_w1, w1tH, E, HD, E * HD);
  k_addvec<<<grid1(1024), 256, 0, stream>>>(lstm_bih, lstm_bhh, lbias, 1024);

  // ---- embed: x = inputs @ embed_w + embed_b (K padded to 320) ----
  gemm16<4><<<dim3(M / 128, E / 128), 256, 0, stream>>>(
      inPadH, WINP, embWt, embed_b, nullptr, xF, nullptr, E, WINP);

  // ---- encoder layers ----
  for (int l = 0; l < NL; ++l) {
    const _Float16* wq = qkvoWt + (size_t)(l * 4 + 0) * E * E;
    const _Float16* wk = qkvoWt + (size_t)(l * 4 + 1) * E * E;
    const _Float16* wv = qkvoWt + (size_t)(l * 4 + 2) * E * E;
    const _Float16* wo = qkvoWt + (size_t)(l * 4 + 3) * E * E;
    const float* bq = qkvo_b + (l * 4 + 0) * E;
    const float* bk = qkvo_b + (l * 4 + 1) * E;
    const float* bv = qkvo_b + (l * 4 + 2) * E;
    const float* bo = qkvo_b + (l * 4 + 3) * E;

    k_layernorm256<<<M / 8, 256, 0, stream>>>(xF, ln_g + (l * 2) * E, ln_b + (l * 2) * E, yH);

    gemm16<8><<<dim3(M / 128, E / 128), 256, 0, stream>>>(yH, E, wq, bq, nullptr, nullptr, qH, E, E);
    gemm16<8><<<dim3(M / 128, E / 128), 256, 0, stream>>>(yH, E, wk, bk, nullptr, nullptr, kH, E, E);
    gemm16<8><<<dim3(M / 128, E / 128), 256, 0, stream>>>(yH, E, wv, bv, nullptr, nullptr, vH, E, E);

    k_attention<<<dim3(Bsz * NH, S / 64), 256, ATT_SMEM, stream>>>(qH, kH, vH, mask, aH);

    // out-proj + residual into x
    gemm16<6><<<dim3(M / 128, E / 128), 256, 0, stream>>>(aH, E, wo, bo, xF, xF, nullptr, E, E);

    k_layernorm256<<<M / 8, 256, 0, stream>>>(xF, ln_g + (l * 2 + 1) * E, ln_b + (l * 2 + 1) * E, yH);

    // ff1 (relu, f16 out), ff2 (+residual into x)
    gemm16<9><<<dim3(M / 128, F / 128), 256, 0, stream>>>(
        yH, E, ffW1t + (size_t)l * F * E, ff_b1 + l * F, nullptr, nullptr, fH, F, E);
    gemm16<6><<<dim3(M / 128, E / 128), 256, 0, stream>>>(
        fH, F, ffW2t + (size_t)l * E * F, ff_b2 + l * E, xF, xF, nullptr, E, F);
  }

  // ---- final encoder LN -> f16 enc ----
  k_layernorm256<<<M / 8, 256, 0, stream>>>(xF, enc_ln_g, enc_ln_b, encH);

  // ---- LSTM decoder scan (single 32-wave workgroup) ----
  k_lstm<<<1, 1024, LSTM_SMEM, stream>>>(encH, mask, wihH, whhH, lbias,
                                         w1tH, out_b1, out_w2, out_b2,
                                         dec_h0, dec_c0, out);
}
